// MultiHeadAttention_61194694034333
// MI455X (gfx1250) — matrix operand order repaired and checked by reading, not compiled, not run on hardware
//
#include <hip/hip_runtime.h>
#include <hip/hip_bf16.h>

// MHA forward for MI455X (gfx1250, wave32, WMMA + TDM).
// Pipeline: cvt(f32->f16) -> QKV GEMMs (f16 WMMA, f32 acc) -> flash-attn -> out-proj.
// Non-transposed LDS tiles are staged by the Tensor Data Mover (tensor_load_to_lds,
// TENSORcnt); transposed tiles (B^T, V^T) use coalesced b128 loads + b16 scatter.

typedef __attribute__((ext_vector_type(16))) _Float16 v16h;
typedef __attribute__((ext_vector_type(8)))  _Float16 v8h;
typedef __attribute__((ext_vector_type(8)))  float    v8f;
typedef __attribute__((ext_vector_type(4)))  unsigned v4u;
typedef __attribute__((ext_vector_type(8)))  int      v8i;
typedef __attribute__((ext_vector_type(4)))  int      v4i;

#define D_MODEL 1024
#define SEQ     2048
#define BATCH   2
#define HEADS   16
#define HDIM    64
#define MTOT    (BATCH*SEQ)   // 4096

#define BM 128
#define BN 64
#define BK 32
#define LDA 40   // LDS stride (halfs) for A tiles, row-major [BM][BK]: 64B + 16B pad
#define LDB 40   // LDS stride (halfs) for B tiles, transposed [BN][BK]

#if defined(__has_builtin)
#if __has_builtin(__builtin_amdgcn_tensor_load_to_lds)
#define USE_TDM 1
#endif
#endif
#ifndef USE_TDM
#define USE_TDM 0
#endif

__device__ inline v8f wmma16(v16h a, v16h b, v8f c) {
  // D(16x16,f32) = A(16x32,f16) * B(32x16,f16) + C
  return __builtin_amdgcn_wmma_f32_16x16x32_f16(false, a, false, b, (short)0, c,
                                                false, false);
}

__device__ inline v16h pack16(v8h lo, v8h hi) {
  v16h r;
#pragma unroll
  for (int e = 0; e < 8; ++e) { r[e] = lo[e]; r[e + 8] = hi[e]; }
  return r;
}

__device__ inline v8f vzero8() {
  v8f z;
#pragma unroll
  for (int i = 0; i < 8; ++i) z[i] = 0.0f;
  return z;
}

#if USE_TDM
// Issue one TDM 2D tile load: global (row-major, rowStrideElem halfs) -> LDS with
// row padding. tileW halfs per row, tileH rows. D# per CDNA5 ISA ch.8:
// padInterval: 0=2,1=4,2=8,3=16,4=32 DWORDs of data before each pad;
// padAmount:   0=1,1=2,...  DWORDs of pad inserted (3 => 16B).
__device__ inline void tdm_load_2d(const _Float16* gsrc, unsigned ldsByteAddr,
                                   unsigned tileW, unsigned tileH,
                                   unsigned rowStrideElem,
                                   unsigned padInterval, unsigned padAmount) {
  unsigned long long ga = (unsigned long long)(uintptr_t)gsrc;
  const unsigned td0 = 1u << 30;   // tensor_dim0 (large: no OOB clipping)
  const unsigned td1 = 1u << 30;   // tensor_dim1

  v4u g0;
  g0.x = 1u;                                   // count=1 (valid descriptor)
  g0.y = ldsByteAddr;                          // lds_addr (bytes)
  g0.z = (unsigned)(ga & 0xffffffffu);         // global_addr[31:0]
  g0.w = (unsigned)((ga >> 32) & 0x01ffffffu)  // global_addr[56:32]
         | (2u << 30);                         // type=2 ("image")

  v8i g1;
  g1[0] = (int)((1u << 16)                     // data_size=1 (2 bytes/elem)
              | (1u << 20)                     // pad_enable
              | (padInterval << 22) | (padAmount << 25));
  g1[1] = (int)((td0 & 0xffffu) << 16);        // [47:32]=atomic addr 0 | dim0 lo
  g1[2] = (int)((td0 >> 16) | ((td1 & 0xffffu) << 16));  // dim0 hi | dim1 lo
  g1[3] = (int)((td1 >> 16) | (tileW << 16));  // dim1 hi | tile_dim0
  g1[4] = (int)tileH;                          // tile_dim1 (tile_dim2 = 0)
  g1[5] = (int)rowStrideElem;                  // tensor_dim0_stride[31:0]
  g1[6] = 0;                                   // stride0 hi | stride1 lo
  g1[7] = 0;                                   // stride1 hi

  v4i z4;
  z4[0] = 0; z4[1] = 0; z4[2] = 0; z4[3] = 0;  // groups 2/3 unused (2D tile)
  v8i z8;
#pragma unroll
  for (int i = 0; i < 8; ++i) z8[i] = 0;
  // 6-arg toolchain form: (g0, g1, g2, g3, extra, cpol)
  __builtin_amdgcn_tensor_load_to_lds(g0, g1, z4, z4, z8, 0);
}
#endif

// ---------------------------------------------------------------- convert
__global__ __launch_bounds__(256)
void cvt_f32_f16_kernel(const float* __restrict__ in, _Float16* __restrict__ out,
                        int n) {
  int i = blockIdx.x * blockDim.x + threadIdx.x;
  int stride = gridDim.x * blockDim.x;
  for (; i < n; i += stride) out[i] = (_Float16)in[i];
}

// ---------------------------------------------------------------- GEMM core
// C[mBase:+128, nBase:+64] = A[4096x1024] * B[1024x1024], f16 in, f32 acc.
// 256 threads = 8 waves; each wave owns a 32x32 micro-tile = 2x2 WMMA tiles.
__device__ inline void gemm_core(const _Float16* __restrict__ A,
                                 const _Float16* __restrict__ B,
                                 _Float16* As, _Float16* Bs,
                                 int mBase, int nBase, v8f (&acc)[2][2]) {
  const int tid  = threadIdx.x;
  const int lane = tid & 31;
  const int wave = tid >> 5;
  const int m0   = (wave & 3) * 32;
  const int n0   = (wave >> 2) * 32;
  const int lh   = lane >> 4;   // lane-half select (ISA fragment layouts)
  const int ll   = lane & 15;

  acc[0][0] = vzero8(); acc[0][1] = vzero8();
  acc[1][0] = vzero8(); acc[1][1] = vzero8();

  for (int kt = 0; kt < D_MODEL / BK; ++kt) {
    __syncthreads();   // protect LDS tiles being overwritten
#if USE_TDM
    // A tile [128 rows x 32 halfs] via Tensor Data Mover, wave 0 issues once.
    if (wave == 0) {
      tdm_load_2d(&A[(size_t)mBase * D_MODEL + kt * BK],
                  (unsigned)(uintptr_t)(void*)As,
                  BK, BM, D_MODEL, /*padInterval=*/3, /*padAmount=*/3);
    }
#else
    for (int i = tid; i < BM * (BK / 8); i += 256) {
      int r = i >> 2, s = i & 3;
      *(v8h*)&As[r * LDA + s * 8] =
          *(const v8h*)&A[(size_t)(mBase + r) * D_MODEL + kt * BK + s * 8];
    }
#endif
    // Stage B tile transposed into [BN][BK]: coalesced b128 loads, b16 scatter.
    for (int i = tid; i < BK * (BN / 8); i += 256) {
      int k = i >> 3, s = i & 7;
      v8h w = *(const v8h*)&B[(size_t)(kt * BK + k) * D_MODEL + nBase + s * 8];
#pragma unroll
      for (int e = 0; e < 8; ++e) Bs[(s * 8 + e) * LDB + k] = w[e];
    }
    if (kt + 1 < D_MODEL / BK)  // L2 hint for next A tile (global_prefetch_b8)
      __builtin_prefetch(&A[(size_t)(mBase + (tid >> 1)) * D_MODEL + (kt + 1) * BK], 0, 0);
#if USE_TDM
    if (wave == 0) __builtin_amdgcn_s_wait_tensorcnt(0);
#endif
    __syncthreads();

    // A fragment: lane ll holds row M=ll; lanes<16: K=0..7 & 16..23; +8 for lh=1.
    v16h af[2];
#pragma unroll
    for (int mi = 0; mi < 2; ++mi) {
      const _Float16* p = &As[(m0 + mi * 16 + ll) * LDA + lh * 8];
      af[mi] = pack16(*(const v8h*)p, *(const v8h*)(p + 16));
    }
    // B fragment: lane ll holds col N=ll; K = lh*8 + 0..7 & 16 + lh*8 + 0..7 (Bs^T).
    v16h bf[2];
#pragma unroll
    for (int ni = 0; ni < 2; ++ni) {
      const _Float16* p = &Bs[(n0 + ni * 16 + ll) * LDB + lh * 8];
      bf[ni] = pack16(*(const v8h*)p, *(const v8h*)(p + 16));
    }
#pragma unroll
    for (int mi = 0; mi < 2; ++mi)
#pragma unroll
      for (int ni = 0; ni < 2; ++ni)
        acc[mi][ni] = wmma16(af[mi], bf[ni], acc[mi][ni]);
  }
}

// ---------------------------------------------------------------- QKV GEMM
// Epilogue scatters f16 results into [B,H,N,dh] layout.
__global__ __launch_bounds__(256)
void qkv_gemm_kernel(const _Float16* __restrict__ A, const _Float16* __restrict__ W,
                     const float* __restrict__ bias, _Float16* __restrict__ Out) {
  __shared__ _Float16 As[BM * LDA];
  __shared__ _Float16 Bs[BN * LDB];
  const int mBase = blockIdx.y * BM;
  const int nBase = blockIdx.x * BN;
  v8f acc[2][2];
  gemm_core(A, W, As, Bs, mBase, nBase, acc);

  const int lane = threadIdx.x & 31;
  const int wave = threadIdx.x >> 5;
  const int m0 = (wave & 3) * 32, n0 = (wave >> 2) * 32;
  const int lh = lane >> 4, ll = lane & 15;
#pragma unroll
  for (int mi = 0; mi < 2; ++mi)
#pragma unroll
    for (int ni = 0; ni < 2; ++ni) {
      int c = nBase + n0 + ni * 16 + ll;        // output feature (C layout N=ll)
      int h = c >> 6, d = c & (HDIM - 1);
      float bv = bias[c];
#pragma unroll
      for (int v = 0; v < 8; ++v) {
        int m = mBase + m0 + mi * 16 + v + 8 * lh;   // C layout: M = v + 8*lh
        int b = m >> 11, n = m & (SEQ - 1);
        Out[((size_t)((b * HEADS + h) * SEQ + n)) * HDIM + d] =
            (_Float16)(acc[mi][ni][v] + bv);
      }
    }
}

// ---------------------------------------------------------------- out-proj GEMM
__global__ __launch_bounds__(256)
void oproj_gemm_kernel(const _Float16* __restrict__ A, const _Float16* __restrict__ W,
                       const float* __restrict__ bias, float* __restrict__ Out) {
  __shared__ _Float16 As[BM * LDA];
  __shared__ _Float16 Bs[BN * LDB];
  const int mBase = blockIdx.y * BM;
  const int nBase = blockIdx.x * BN;
  v8f acc[2][2];
  gemm_core(A, W, As, Bs, mBase, nBase, acc);

  const int lane = threadIdx.x & 31;
  const int wave = threadIdx.x >> 5;
  const int m0 = (wave & 3) * 32, n0 = (wave >> 2) * 32;
  const int lh = lane >> 4, ll = lane & 15;
#pragma unroll
  for (int mi = 0; mi < 2; ++mi)
#pragma unroll
    for (int ni = 0; ni < 2; ++ni) {
      int c = nBase + n0 + ni * 16 + ll;
      float bv = bias[c];
#pragma unroll
      for (int v = 0; v < 8; ++v) {
        int m = mBase + m0 + mi * 16 + v + 8 * lh;
        Out[(size_t)m * D_MODEL + c] = acc[mi][ni][v] + bv;
      }
    }
}

// ---------------------------------------------------------------- attention
// Grid (32 q-tiles, 32 b*h). Block 128 = 4 waves; wave owns 16 query rows.
// Flash-attention over key blocks of 32; f16 WMMA with f32 online softmax.
__global__ __launch_bounds__(128)
void attn_kernel(const _Float16* __restrict__ Q, const _Float16* __restrict__ K,
                 const _Float16* __restrict__ V, _Float16* __restrict__ ctx) {
  __shared__ _Float16 Ks[32 * 72];       // K tile row-major [key][d]: 128B + 16B pad
  __shared__ _Float16 Vt[64 * 40];       // V tile transposed [d][key], pad 40
  __shared__ _Float16 Ps[4 * 16 * 40];   // per-wave P staging 16x32, pad 40

  const int tid  = threadIdx.x;
  const int lane = tid & 31;
  const int wave = tid >> 5;
  const int lh = lane >> 4, ll = lane & 15;

  const int bh = blockIdx.y;                     // b*16 + h
  const size_t headOff = (size_t)bh * SEQ * HDIM;
  const _Float16* Qh = Q + headOff;
  const _Float16* Kh = K + headOff;
  const _Float16* Vh = V + headOff;
  const int qbase = blockIdx.x * 64 + wave * 16;
  _Float16* Pw = &Ps[wave * 16 * 40];

  // Q fragments for the wave's 16 rows: two d-chunks of 32 (A-frag layout).
  v16h qf[2];
#pragma unroll
  for (int dc = 0; dc < 2; ++dc) {
    const _Float16* p = &Qh[(size_t)(qbase + ll) * HDIM + dc * 32 + lh * 8];
    qf[dc] = pack16(*(const v8h*)p, *(const v8h*)(p + 16));
  }

  v8f o[4];  // 16x64 output accumulator (4 C tiles along d)
#pragma unroll
  for (int t = 0; t < 4; ++t) o[t] = vzero8();
  float mrow[8], lrow[8];
#pragma unroll
  for (int v = 0; v < 8; ++v) { mrow[v] = -1e30f; lrow[v] = 0.0f; }

  for (int kb = 0; kb < SEQ / 32; ++kb) {
    __syncthreads();
#if USE_TDM
    // K tile [32 keys x 64 halfs] via Tensor Data Mover (rows 128B + 16B pad).
    if (wave == 0) {
      tdm_load_2d(&Kh[(size_t)(kb * 32) * HDIM],
                  (unsigned)(uintptr_t)(void*)Ks,
                  HDIM, 32, HDIM, /*padInterval=*/4, /*padAmount=*/3);
    }
#else
    for (int i = tid; i < 32 * 8; i += 128) {
      int key = i >> 3, s = i & 7;
      *(v8h*)&Ks[key * 72 + s * 8] =
          *(const v8h*)&Kh[(size_t)(kb * 32 + key) * HDIM + s * 8];
    }
#endif
    // Stage V tile transposed (b128 loads, b16 scatter).
    for (int i = tid; i < 32 * 8; i += 128) {
      int key = i >> 3, s = i & 7;
      v8h w = *(const v8h*)&Vh[(size_t)(kb * 32 + key) * HDIM + s * 8];
#pragma unroll
      for (int e = 0; e < 8; ++e) Vt[(s * 8 + e) * 40 + key] = w[e];
    }
#if USE_TDM
    if (wave == 0) __builtin_amdgcn_s_wait_tensorcnt(0);
#endif
    __syncthreads();

    // Scores S[16 q x 32 key] = Q * K^T : two 16x16 C tiles, K-dim = dh = 64.
    v8f sc[2];
    sc[0] = vzero8(); sc[1] = vzero8();
#pragma unroll
    for (int j = 0; j < 2; ++j)
#pragma unroll
      for (int dc = 0; dc < 2; ++dc) {
        // B[d][key]: lane holds key=j*16+ll, d = dc*32 + lh*8 + 0..7 & +16.
        const _Float16* p = &Ks[(j * 16 + ll) * 72 + dc * 32 + lh * 8];
        v16h bf = pack16(*(const v8h*)p, *(const v8h*)(p + 16));
        sc[j] = wmma16(qf[dc], bf, sc[j]);
      }

    // Online softmax per row (row = v + 8*lh; elements across 16-lane half).
#pragma unroll
    for (int v = 0; v < 8; ++v) {
      float x0 = sc[0][v] * 0.125f;   // 1/sqrt(64)
      float x1 = sc[1][v] * 0.125f;
      float mx = fmaxf(x0, x1);
#pragma unroll
      for (int msk = 8; msk >= 1; msk >>= 1) mx = fmaxf(mx, __shfl_xor(mx, msk));
      float mnew = fmaxf(mrow[v], mx);
      float corr = __expf(mrow[v] - mnew);
      float p0 = __expf(x0 - mnew);
      float p1 = __expf(x1 - mnew);
      float rs = p0 + p1;
#pragma unroll
      for (int msk = 8; msk >= 1; msk >>= 1) rs += __shfl_xor(rs, msk);
      lrow[v] = lrow[v] * corr + rs;
      mrow[v] = mnew;
#pragma unroll
      for (int t = 0; t < 4; ++t) o[t][v] *= corr;
      int r = v + 8 * lh;                       // C-layout row within wave tile
      Pw[r * 40 + ll]      = (_Float16)p0;      // stage P (C layout -> LDS)
      Pw[r * 40 + 16 + ll] = (_Float16)p1;
    }
    __syncthreads();

    // Reload P as an A-fragment (16x32) and accumulate O += P * V.
    {
      const _Float16* p = &Pw[ll * 40 + lh * 8];
      v16h pf = pack16(*(const v8h*)p, *(const v8h*)(p + 16));
#pragma unroll
      for (int t = 0; t < 4; ++t) {
        // B[key][d]: lane holds d = t*16+ll, key = lh*8 + 0..7 & 16 + lh*8 + 0..7.
        const _Float16* bp = &Vt[(t * 16 + ll) * 40 + lh * 8];
        v16h bf = pack16(*(const v8h*)bp, *(const v8h*)(bp + 16));
        o[t] = wmma16(pf, bf, o[t]);
      }
    }
  }

  // Finalize: divide by row sums, write merged-heads ctx [B,N,H*dh] as f16.
  const int b = bh >> 4, h = bh & 15;
#pragma unroll
  for (int v = 0; v < 8; ++v) {
    float inv = 1.0f / lrow[v];
    int qrow = qbase + v + 8 * lh;
#pragma unroll
    for (int t = 0; t < 4; ++t) {
      int d = t * 16 + ll;
      ctx[(size_t)(b * SEQ + qrow) * D_MODEL + h * HDIM + d] =
          (_Float16)(o[t][v] * inv);
    }
  }
}

// ---------------------------------------------------------------- launch
extern "C" void kernel_launch(void* const* d_in, const int* in_sizes, int n_in,
                              void* d_out, int out_size, void* d_ws, size_t ws_size,
                              hipStream_t stream) {
  const float* x  = (const float*)d_in[0];
  const float* Wq = (const float*)d_in[1];
  const float* bq = (const float*)d_in[2];
  const float* Wk = (const float*)d_in[3];
  const float* bk = (const float*)d_in[4];
  const float* Wv = (const float*)d_in[5];
  const float* bv = (const float*)d_in[6];
  const float* Wo = (const float*)d_in[7];
  const float* bo = (const float*)d_in[8];
  float* out = (float*)d_out;

  char* ws = (char*)d_ws;
  const size_t MiB = 1024ull * 1024ull;
  _Float16* x16  = (_Float16*)(ws + 0 * MiB);   // 8 MiB
  _Float16* Wq16 = (_Float16*)(ws + 8 * MiB);   // 2 MiB
  _Float16* Wk16 = (_Float16*)(ws + 10 * MiB);
  _Float16* Wv16 = (_Float16*)(ws + 12 * MiB);
  _Float16* Wo16 = (_Float16*)(ws + 14 * MiB);
  _Float16* Q16  = (_Float16*)(ws + 16 * MiB);  // [B,H,N,dh] 8 MiB
  _Float16* K16  = (_Float16*)(ws + 24 * MiB);
  _Float16* V16  = (_Float16*)(ws + 32 * MiB);
  _Float16* C16  = (_Float16*)(ws + 40 * MiB);  // merged ctx [B,N,D] 8 MiB

  cvt_f32_f16_kernel<<<4096, 256, 0, stream>>>(x,  x16,  MTOT * D_MODEL);
  cvt_f32_f16_kernel<<<1024, 256, 0, stream>>>(Wq, Wq16, D_MODEL * D_MODEL);
  cvt_f32_f16_kernel<<<1024, 256, 0, stream>>>(Wk, Wk16, D_MODEL * D_MODEL);
  cvt_f32_f16_kernel<<<1024, 256, 0, stream>>>(Wv, Wv16, D_MODEL * D_MODEL);
  cvt_f32_f16_kernel<<<1024, 256, 0, stream>>>(Wo, Wo16, D_MODEL * D_MODEL);

  dim3 gg(D_MODEL / BN, MTOT / BM);   // (16, 32)
  qkv_gemm_kernel<<<gg, 256, 0, stream>>>(x16, Wq16, bq, Q16);
  qkv_gemm_kernel<<<gg, 256, 0, stream>>>(x16, Wk16, bk, K16);
  qkv_gemm_kernel<<<gg, 256, 0, stream>>>(x16, Wv16, bv, V16);

  attn_kernel<<<dim3(SEQ / 64, BATCH * HEADS), 128, 0, stream>>>(Q16, K16, V16, C16);

  oproj_gemm_kernel<<<gg, 256, 0, stream>>>(C16, Wo16, bo, out);
}
